// GPT_65919158059439
// MI455X (gfx1250) — compile-verified, matrix operand order checked by reading, not run on hardware
//
#include <hip/hip_runtime.h>
#include <hip/hip_bf16.h>
#include <math.h>

typedef __attribute__((ext_vector_type(16))) _Float16 v16h;
typedef __attribute__((ext_vector_type(8)))  float    v8f;

namespace {
constexpr int kV = 50257;
constexpr int kE = 1024;
constexpr int kH = 16;
constexpr int kL = 6;
constexpr int kT = 1024;
constexpr int kB = 2;
constexpr int kD = 64;          // head dim
constexpr int kMT = kB * kT;    // 2048 activation rows
constexpr int WIN = 256;
constexpr float SOFTCAP = 30.0f;

constexpr int BM = 128, BN = 128, BK = 32, LDH = BK + 8;  // 40 halves = 80B rows

union Frag16 { v16h v; unsigned u[8]; _Float16 h[16]; };

// Per-lane K offset pattern for 16-bit A/B fragments (ISA 7.12.2):
// VGPR v holds K = {base,base+1}; base = (v<4?0:16) + 8*halfgroup + 2*(v&3)
__device__ __forceinline__ int koff(int v, int hlf) {
  return ((v < 4) ? 0 : 16) + hlf * 8 + (v & 3) * 2;
}

__device__ __forceinline__ v8f wmma_f16(v16h a, v16h b, v8f c) {
  return __builtin_amdgcn_wmma_f32_16x16x32_f16(false, a, false, b, (short)0, c,
                                                false, false);
}

__device__ __forceinline__ unsigned pack2h(float a, float b) {
  union { _Float16 h[2]; unsigned u; } t;
  t.h[0] = (_Float16)a; t.h[1] = (_Float16)b;
  return t.u;
}

// Raw 32-bit LDS offset of a __shared__ pointer (AS3 offset == low 32 addr bits)
__device__ __forceinline__ unsigned lds_off(const void* p) {
  return (unsigned)(unsigned long long)p;
}

// Async global->LDS copy, 16B per active lane. ASYNCcnt-tracked.
__device__ __forceinline__ void async_copy_b128(unsigned lds, const void* gaddr) {
  asm volatile("global_load_async_to_lds_b128 %0, %1, off"
               :: "v"(lds), "v"(gaddr) : "memory");
}
__device__ __forceinline__ void wait_async0() {
  asm volatile("s_wait_asynccnt 0" ::: "memory");
}
} // namespace

// ---------------------------------------------------------------------------
// Embedding: x[b,t,:] = wte[idx[b,t],:] + wpe[t,:]
// ---------------------------------------------------------------------------
__global__ void __launch_bounds__(256)
embed_kernel(const int* __restrict__ idx, const float* __restrict__ wte,
             const float* __restrict__ wpe, float* __restrict__ x) {
  const int bt = blockIdx.x;
  const int t = bt % kT;
  const int tok = idx[bt];
  const float* wr = wte + (size_t)tok * kE;
  const float* pr = wpe + (size_t)t * kE;
  float* xo = x + (size_t)bt * kE;
  for (int i = threadIdx.x; i < kE; i += 256) xo[i] = wr[i] + pr[i];
}

// ---------------------------------------------------------------------------
// LayerNorm: one row (E=1024) per 256-thread block. OutT = _Float16 for the
// GEMM-feeding path (enables async WMMA staging), float for the logits path.
// ---------------------------------------------------------------------------
template <typename OutT>
__global__ void __launch_bounds__(256)
ln_kernel(const float* __restrict__ in, const float* __restrict__ w,
          const float* __restrict__ b, OutT* __restrict__ out,
          size_t in_stride, size_t out_stride) {
  const int tid = threadIdx.x;
  const float* xr = in + (size_t)blockIdx.x * in_stride;
  float vbuf[4];
  float s = 0.f, s2 = 0.f;
#pragma unroll
  for (int i = 0; i < 4; ++i) {
    float v = xr[tid + i * 256];
    vbuf[i] = v; s += v; s2 += v * v;
  }
#pragma unroll
  for (int m = 16; m >= 1; m >>= 1) {
    s += __shfl_xor(s, m, 32);
    s2 += __shfl_xor(s2, m, 32);
  }
  __shared__ float rs[8], rs2[8];
  __shared__ float smu, srstd;
  if ((tid & 31) == 0) { rs[tid >> 5] = s; rs2[tid >> 5] = s2; }
  __syncthreads();
  if (tid == 0) {
    float a = 0.f, c = 0.f;
#pragma unroll
    for (int i = 0; i < 8; ++i) { a += rs[i]; c += rs2[i]; }
    float mu = a * (1.0f / (float)kE);
    smu = mu;
    srstd = rsqrtf(c * (1.0f / (float)kE) - mu * mu + 1e-5f);
  }
  __syncthreads();
  const float mu = smu, rstd = srstd;
  OutT* orow = out + (size_t)blockIdx.x * out_stride;
#pragma unroll
  for (int i = 0; i < 4; ++i) {
    int e = tid + i * 256;
    orow[e] = (OutT)((vbuf[i] - mu) * rstd * w[e] + b[e]);
  }
}

// ---------------------------------------------------------------------------
// WMMA GEMM: C[M,N] = epilogue(A[M,K] @ B[K,N] + bias[N])
//   EPI 0: none      EPI 1: += extra[m,n] (residual)
//   EPI 2: *= silu(extra[m,n])   (SiLU-GLU gate)
// AT = _Float16: A-tile staged with global_load_async_to_lds_b128 (no convert)
// AT = float   : A-tile converted fp32->f16 during staging.
// 128x128 block tile, 8 waves in 4(M)x2(N) grid, 32x64 per wave = 8 WMMA accs.
// Dims assumed multiples of the tiling (true for this model).
// ---------------------------------------------------------------------------
template <int EPI, typename AT, typename OutT>
__global__ void __launch_bounds__(256)
gemm_kernel(const AT* __restrict__ A, const float* __restrict__ Bp,
            const float* __restrict__ bias, const float* __restrict__ extra,
            OutT* __restrict__ C, int M, int N, int K) {
  (void)M;
  __shared__ __align__(16) _Float16 As[BM][LDH];   // [m][k]
  __shared__ __align__(16) _Float16 Bs[BN][LDH];   // [n][k] (transposed staging)
  const int tid = threadIdx.x;
  const int lane = tid & 31, wid = tid >> 5;
  const int hlf = lane >> 4, idx = lane & 15;
  const int wm = (wid & 3) * 32, wn = (wid >> 2) * 64;
  const int m0 = blockIdx.y * BM, n0 = blockIdx.x * BN;

  const v8f vzero = {0.f, 0.f, 0.f, 0.f, 0.f, 0.f, 0.f, 0.f};
  v8f acc[2][4];
#pragma unroll
  for (int i = 0; i < 2; ++i)
#pragma unroll
    for (int j = 0; j < 4; ++j) acc[i][j] = vzero;

  for (int k0 = 0; k0 < K; k0 += BK) {
    if constexpr (sizeof(AT) == 2) {
      // Async DMA: 128 rows x 64B; 512 16B chunks over 256 threads.
#pragma unroll
      for (int it = 0; it < 2; ++it) {
        int e = tid + it * 256;
        int r = e >> 2, c8 = (e & 3) * 8;
        async_copy_b128(lds_off(&As[r][c8]),
                        A + (size_t)(m0 + r) * K + (k0 + c8));
      }
    } else {
      // fp32 -> f16 convert while staging, float4 per thread per pass.
#pragma unroll
      for (int it = 0; it < 4; ++it) {
        int r = it * 32 + (tid >> 3);
        int c = (tid & 7) * 4;
        float4 f = *(const float4*)(A + (size_t)(m0 + r) * K + (k0 + c));
        *(unsigned*)&As[r][c]     = pack2h(f.x, f.y);
        *(unsigned*)&As[r][c + 2] = pack2h(f.z, f.w);
      }
    }
    // Stage B tile transposed to [n][k] (fp32 weights -> f16).
#pragma unroll
    for (int it = 0; it < 4; ++it) {
      int kk = it * 8 + (tid >> 5);
      int c = (tid & 31) * 4;
      float4 f = *(const float4*)(Bp + (size_t)(k0 + kk) * N + (n0 + c));
      Bs[c + 0][kk] = (_Float16)f.x;
      Bs[c + 1][kk] = (_Float16)f.y;
      Bs[c + 2][kk] = (_Float16)f.z;
      Bs[c + 3][kk] = (_Float16)f.w;
    }
    if (k0 + BK < K)
      __builtin_prefetch(Bp + (size_t)(k0 + BK + (tid >> 5)) * N + n0, 0, 1);
    if constexpr (sizeof(AT) == 2) wait_async0();
    __syncthreads();

    Frag16 af[2], bf[4];
#pragma unroll
    for (int mt = 0; mt < 2; ++mt)
#pragma unroll
      for (int v = 0; v < 8; ++v)
        af[mt].u[v] = *(const unsigned*)&As[wm + mt * 16 + idx][koff(v, hlf)];
#pragma unroll
    for (int nt = 0; nt < 4; ++nt)
#pragma unroll
      for (int v = 0; v < 8; ++v)
        bf[nt].u[v] = *(const unsigned*)&Bs[wn + nt * 16 + idx][koff(v, hlf)];
#pragma unroll
    for (int mt = 0; mt < 2; ++mt)
#pragma unroll
      for (int nt = 0; nt < 4; ++nt)
        acc[mt][nt] = wmma_f16(af[mt].v, bf[nt].v, acc[mt][nt]);
    __syncthreads();
  }

  // Epilogue. C/D layout: element r -> row M = r + 8*halfgroup, col N = idx.
#pragma unroll
  for (int mt = 0; mt < 2; ++mt)
#pragma unroll
    for (int nt = 0; nt < 4; ++nt)
#pragma unroll
      for (int r = 0; r < 8; ++r) {
        int m = m0 + wm + mt * 16 + hlf * 8 + r;
        int n = n0 + wn + nt * 16 + idx;
        float val = acc[mt][nt][r] + bias[n];
        if (EPI == 1) val += extra[(size_t)m * N + n];
        if (EPI == 2) {
          float u = extra[(size_t)m * N + n];
          val *= u / (1.0f + __expf(-u));
        }
        C[(size_t)m * N + n] = (OutT)val;
      }
}

// ---------------------------------------------------------------------------
// Flash attention with WMMA on f16 qkv ([B,T,3E], head h at h*D).
// One block (4 waves) = 64 query rows of one (batch, head); 16 rows per wave.
// K tiles staged via async DMA; V staged with an LDS transpose; S = Q K^T
// (2 WMMA) -> scale+softcap+mask -> online softmax -> P V (4 WMMA).
// Mask: (causal & q-k<=256) OR (k < suffix_prefix_length[b]).
// ---------------------------------------------------------------------------
__global__ void __launch_bounds__(128)
attn_kernel(const _Float16* __restrict__ qkv, const int* __restrict__ spl,
            float* __restrict__ y) {
  __shared__ __align__(16) _Float16 Ks[32][kD + 8];    // [kv][d], 144B rows
  __shared__ __align__(16) _Float16 Vs[kD][32 + 8];    // [d][kv]
  __shared__ __align__(16) _Float16 Ps[4][16][32 + 8]; // per-wave P transpose
  const int tid = threadIdx.x;
  const int lane = tid & 31, wid = tid >> 5;
  const int hlf = lane >> 4, idx = lane & 15;
  const int qblk = blockIdx.x, hh = blockIdx.y, bb = blockIdx.z;
  const int qmin = qblk * 64, qmax = qmin + 63;
  const int q0 = qmin + wid * 16;
  const int sp = spl[bb];
  const float scale = 0.125f;  // 1/sqrt(D), applied post-WMMA

  // Q fragments: A-layout, row = idx; raw packed-half copies.
  Frag16 qf[2];
#pragma unroll
  for (int kh = 0; kh < 2; ++kh) {
    size_t base = ((size_t)(bb * kT + (q0 + idx)) * 3 * kE) + (size_t)hh * kD;
#pragma unroll
    for (int v = 0; v < 8; ++v) {
      int d = kh * 32 + koff(v, hlf);
      qf[kh].u[v] = *(const unsigned*)(qkv + base + d);
    }
  }

  const float NEG = -__builtin_inff();
  const v8f vzero = {0.f, 0.f, 0.f, 0.f, 0.f, 0.f, 0.f, 0.f};
  float mrow[8], lrow[8];
  v8f yacc[4];
#pragma unroll
  for (int r = 0; r < 8; ++r) { mrow[r] = NEG; lrow[r] = 0.f; }
#pragma unroll
  for (int nt = 0; nt < 4; ++nt) yacc[nt] = vzero;

  for (int kb0 = 0; kb0 < kT; kb0 += 32) {
    // Block-uniform skip: no causal-window overlap and no prefix coverage.
    bool any = ((kb0 <= qmax) && (kb0 + 31 >= qmin - WIN)) || (kb0 < sp);
    if (!any) continue;

    // K tile: 32 rows x 128B via async DMA (256 chunks over 128 threads).
#pragma unroll
    for (int it = 0; it < 2; ++it) {
      int e = tid + it * 128;
      int kv = e >> 3, c8 = (e & 7) * 8;
      const _Float16* g = qkv + (size_t)(bb * kT + kb0 + kv) * 3 * kE + kE +
                          (size_t)hh * kD + c8;
      async_copy_b128(lds_off(&Ks[kv][c8]), g);
    }
    // V tile: transpose into [d][kv] (4 halves per thread per pass).
#pragma unroll
    for (int it = 0; it < 4; ++it) {
      int e = tid + it * 128;
      int kv = e >> 4, d4 = (e & 15) * 4;
      union { uint2 u2; _Float16 h[4]; } t;
      t.u2 = *(const uint2*)(qkv + (size_t)(bb * kT + kb0 + kv) * 3 * kE +
                             2 * kE + (size_t)hh * kD + d4);
      Vs[d4 + 0][kv] = t.h[0];
      Vs[d4 + 1][kv] = t.h[1];
      Vs[d4 + 2][kv] = t.h[2];
      Vs[d4 + 3][kv] = t.h[3];
    }
    wait_async0();
    __syncthreads();

    // Scores for two 16-key tiles.
    float sraw[2][8];
#pragma unroll
    for (int kt = 0; kt < 2; ++kt) {
      Frag16 kf0, kf1;  // B-layout: col = kv idx, K pattern over d
#pragma unroll
      for (int v = 0; v < 8; ++v) {
        kf0.u[v] = *(const unsigned*)&Ks[kt * 16 + idx][koff(v, hlf)];
        kf1.u[v] = *(const unsigned*)&Ks[kt * 16 + idx][32 + koff(v, hlf)];
      }
      v8f st = vzero;
      st = wmma_f16(qf[0].v, kf0.v, st);
      st = wmma_f16(qf[1].v, kf1.v, st);
#pragma unroll
      for (int r = 0; r < 8; ++r) {
        float sv = st[r] * scale;
        sv = SOFTCAP * tanhf(sv * (1.0f / SOFTCAP));
        int qg = q0 + hlf * 8 + r;
        int kg = kb0 + kt * 16 + idx;
        bool valid = ((qg >= kg) && (qg - kg <= WIN)) || (kg < sp);
        sraw[kt][r] = valid ? sv : NEG;
      }
    }

    // Online softmax update; row stats live across the 16 lanes of a half.
    float alpha[8];
#pragma unroll
    for (int r = 0; r < 8; ++r) {
      float rm = fmaxf(sraw[0][r], sraw[1][r]);
#pragma unroll
      for (int m = 8; m >= 1; m >>= 1) rm = fmaxf(rm, __shfl_xor(rm, m, 32));
      float mn = fmaxf(mrow[r], rm);
      float al = (mn == NEG) ? 1.0f : __expf(mrow[r] - mn);
      float ps = 0.f;
#pragma unroll
      for (int kt = 0; kt < 2; ++kt) {
        float sv = sraw[kt][r];
        float p = (sv == NEG) ? 0.f : __expf(sv - mn);
        ps += p;
        Ps[wid][hlf * 8 + r][kt * 16 + idx] = (_Float16)p;
      }
#pragma unroll
      for (int m = 8; m >= 1; m >>= 1) ps += __shfl_xor(ps, m, 32);
      lrow[r] = lrow[r] * al + ps;
      mrow[r] = mn;
      alpha[r] = al;
    }
#pragma unroll
    for (int nt = 0; nt < 4; ++nt)
#pragma unroll
      for (int r = 0; r < 8; ++r) yacc[nt][r] *= alpha[r];

    // P (A-layout via LDS transpose) x V (B-layout) -> y accumulators.
    Frag16 pf;
#pragma unroll
    for (int v = 0; v < 8; ++v)
      pf.u[v] = *(const unsigned*)&Ps[wid][idx][koff(v, hlf)];
#pragma unroll
    for (int nt = 0; nt < 4; ++nt) {
      Frag16 vf;
#pragma unroll
      for (int v = 0; v < 8; ++v)
        vf.u[v] = *(const unsigned*)&Vs[nt * 16 + idx][koff(v, hlf)];
      yacc[nt] = wmma_f16(pf.v, vf.v, yacc[nt]);
    }
    __syncthreads();
  }

  // Normalize and write y[b, q, h*D + d] (row q always has k=q valid => l>0).
#pragma unroll
  for (int nt = 0; nt < 4; ++nt)
#pragma unroll
    for (int r = 0; r < 8; ++r) {
      int qg = q0 + hlf * 8 + r;
      float l = lrow[r];
      float val = (l > 0.f) ? yacc[nt][r] / l : 0.f;
      y[(size_t)(bb * kT + qg) * kE + hh * kD + nt * 16 + idx] = val;
    }
}

// ---------------------------------------------------------------------------
// Logits for the last position: out[b,v] = dot(xlast[b,:], wte[v,:]).
// Memory-bound on wte; one thread per (b, v), float4 inner loop.
// ---------------------------------------------------------------------------
__global__ void __launch_bounds__(256)
logits_kernel(const float* __restrict__ xl, const float* __restrict__ wte,
              float* __restrict__ out) {
  int n = blockIdx.x * 256 + threadIdx.x;
  if (n >= kB * kV) return;
  int b = n / kV, v = n - b * kV;
  const float4* xr = (const float4*)(xl + (size_t)b * kE);
  const float4* wr = (const float4*)(wte + (size_t)v * kE);
  float acc = 0.f;
#pragma unroll 8
  for (int i = 0; i < kE / 4; ++i) {
    float4 a = xr[i], w = wr[i];
    acc += a.x * w.x + a.y * w.y + a.z * w.z + a.w * w.w;
  }
  out[n] = acc;
}

// ---------------------------------------------------------------------------
extern "C" void kernel_launch(void* const* d_in, const int* in_sizes, int n_in,
                              void* d_out, int out_size, void* d_ws,
                              size_t ws_size, hipStream_t stream) {
  (void)in_sizes; (void)n_in; (void)out_size; (void)ws_size;
  const int*   idx   = (const int*)d_in[0];
  const int*   spl   = (const int*)d_in[1];
  const float* wte   = (const float*)d_in[2];
  const float* wpe   = (const float*)d_in[3];
  const float* ln1w  = (const float*)d_in[4];
  const float* ln1b  = (const float*)d_in[5];
  const float* Wqkv  = (const float*)d_in[6];
  const float* bqkv  = (const float*)d_in[7];
  const float* Wo    = (const float*)d_in[8];
  const float* bo    = (const float*)d_in[9];
  const float* ln2w  = (const float*)d_in[10];
  const float* ln2b  = (const float*)d_in[11];
  const float* Wfc   = (const float*)d_in[12];
  const float* bfc   = (const float*)d_in[13];
  const float* Wfc2  = (const float*)d_in[14];
  const float* bfc2  = (const float*)d_in[15];
  const float* Wproj = (const float*)d_in[16];
  const float* bproj = (const float*)d_in[17];
  const float* lnfw  = (const float*)d_in[18];
  const float* lnfb  = (const float*)d_in[19];
  float* out = (float*)d_out;

  // Workspace layout; regions sized in fp32 units (f16 buffers use half).
  float* x  = (float*)d_ws;                        // [2048,1024] f32 residual
  _Float16* h    = (_Float16*)(x + (size_t)kMT * kE);         // [2048,1024] f16
  _Float16* qkvb = (_Float16*)(x + 2 * (size_t)kMT * kE);     // [2048,3072] f16
  float* yb = x + 2 * (size_t)kMT * kE + (size_t)kMT * 3 * kE / 2 + (size_t)kMT * kE;
  yb        = x + 4 * (size_t)kMT * kE;            // [2048,1024] f32 attn out
  float* ub = yb + (size_t)kMT * kE;               // [2048,4096] f32 fc
  float* gb = ub + (size_t)kMT * 4 * kE;           // [2048,4096] f32 gated
  float* xl = gb + (size_t)kMT * 4 * kE;           // [2,1024] f32 last-pos LN

  embed_kernel<<<kMT, 256, 0, stream>>>(idx, wte, wpe, x);

  for (int l = 0; l < kL; ++l) {
    ln_kernel<_Float16><<<kMT, 256, 0, stream>>>(
        x, ln1w + (size_t)l * kE, ln1b + (size_t)l * kE, h, kE, kE);
    gemm_kernel<0, _Float16, _Float16>
        <<<dim3(3 * kE / BN, kMT / BM), 256, 0, stream>>>(
            h, Wqkv + (size_t)l * kE * 3 * kE, bqkv + (size_t)l * 3 * kE,
            nullptr, qkvb, kMT, 3 * kE, kE);
    attn_kernel<<<dim3(kT / 64, kH, kB), 128, 0, stream>>>(qkvb, spl, yb);
    gemm_kernel<1, float, float><<<dim3(kE / BN, kMT / BM), 256, 0, stream>>>(
        yb, Wo + (size_t)l * kE * kE, bo + (size_t)l * kE, x, x, kMT, kE, kE);
    ln_kernel<_Float16><<<kMT, 256, 0, stream>>>(
        x, ln2w + (size_t)l * kE, ln2b + (size_t)l * kE, h, kE, kE);
    gemm_kernel<0, _Float16, float>
        <<<dim3(4 * kE / BN, kMT / BM), 256, 0, stream>>>(
            h, Wfc + (size_t)l * kE * 4 * kE, bfc + (size_t)l * 4 * kE,
            nullptr, ub, kMT, 4 * kE, kE);
    gemm_kernel<2, _Float16, float>
        <<<dim3(4 * kE / BN, kMT / BM), 256, 0, stream>>>(
            h, Wfc2 + (size_t)l * kE * 4 * kE, bfc2 + (size_t)l * 4 * kE, ub,
            gb, kMT, 4 * kE, kE);
    gemm_kernel<1, float, float><<<dim3(kE / BN, kMT / BM), 256, 0, stream>>>(
        gb, Wproj + (size_t)l * 4 * kE * kE, bproj + (size_t)l * kE, x, x,
        kMT, kE, 4 * kE);
  }

  // Final LN on last position of each batch row, then logits vs wte^T.
  ln_kernel<float><<<kB, 256, 0, stream>>>(x + (size_t)(kT - 1) * kE, lnfw,
                                           lnfb, xl, (size_t)kT * kE, kE);
  logits_kernel<<<(kB * kV + 255) / 256, 256, 0, stream>>>(xl, wte, out);
}
